// Net_86895778332672
// MI455X (gfx1250) — matrix operand order repaired and checked by reading, not compiled, not run on hardware
//
#include <hip/hip_runtime.h>
#include <hip/hip_bf16.h>
#include <math.h>

// ---------------------------------------------------------------------------
// Problem constants (shapes fixed by the reference; N/E derived from in_sizes)
//   F = 42 features, H = 20 LSTM hidden per direction
// ---------------------------------------------------------------------------
#define FEAT 42
#define HID 20
#define H4 80     // 4*H  (== 5 N-tiles exactly)
#define G3 126    // 3*F
#define KPAD 64   // K=42 zero-padded to 2 WMMA k-steps of 32
#define NP_P 48   // padded stride for P1/P2   (3 tiles)
#define NP_G 128  // padded stride for gi/gh   (8 tiles)

typedef __attribute__((ext_vector_type(16))) _Float16 v16h;
typedef __attribute__((ext_vector_type(8)))  _Float16 v8h;
typedef __attribute__((ext_vector_type(8)))  float    v8f;

__device__ __forceinline__ float sigf(float x) {
    return 1.0f / (1.0f + __expf(-x));
}

// ---------------------------------------------------------------------------
// Zero-fill (agg must be zeroed every call: harness does not re-poison)
// ---------------------------------------------------------------------------
__global__ void zero_f32(float* __restrict__ p, int n) {
    int i = blockIdx.x * blockDim.x + threadIdx.x;
    if (i < n) p[i] = 0.0f;
}

// Pad a bias vector to np floats (zero-filled tail); np <= 128, one block.
__global__ void pack_bias(const float* __restrict__ b, int n,
                          float* __restrict__ bp, int np) {
    int i = threadIdx.x;
    if (i < np) bp[i] = (i < n) ? b[i] : 0.0f;
}

// ---------------------------------------------------------------------------
// Operand packing: f32 [M,K] (leading dim ldw) -> f16 [Mp, 64] zero-padded.
// Coalesced; removes ALL bounds checks / predication from the WMMA kernel.
// ---------------------------------------------------------------------------
__global__ void pack_f16(const float* __restrict__ A, int ldw, int M, int K,
                         _Float16* __restrict__ Ap, int Mp) {
    int i = blockIdx.x * blockDim.x + threadIdx.x;
    if (i >= Mp * KPAD) return;
    int r = i >> 6;         // row
    int k = i & (KPAD - 1); // col
    float v = (r < M && k < K) ? A[r * ldw + k] : 0.0f;
    Ap[i] = (_Float16)v;
}

// ---------------------------------------------------------------------------
// WMMA GEMM on packed operands:  C[Mp, Np] = Ap @ Bpᵀ (+biasP)
//   Ap: [Mtiles*16, 64] f16,  Bp: [Ntiles*16, 64] f16 (Bp[n][k] = B[k][n]),
//   biasP: Np floats (zero-padded), C: padded row stride Np = Ntiles*16.
// One wave32 per M-tile; A fragments (both k-steps) loaded once as 4x b128
// and reused across all N-tiles.  Inner loop per N-tile is branch-free:
//   2x 32B B loads + 1 bias load -> 2x v_wmma_f32_16x16x32_f16 -> 8 stores.
//
// Fragment layouts per CDNA5 ISA §7.12.2 (wave32):
//  A: lane m=lane%16; halves h=0..7  -> k = kb + 8*(lane/16) + h
//                     halves h=8..15 -> k = kb + 16 + 8*(lane/16) + (h-8)
//  B: lane n=lane%16; same k mapping as A (halves 0..7 / 8..15 split)
//  C/D: lane n=lane%16; vgpr v -> m = v + 8*(lane/16)
// ---------------------------------------------------------------------------
__global__ void gemm_wmma_f16p(const _Float16* __restrict__ Ap,
                               const _Float16* __restrict__ Bp,
                               const float* __restrict__ biasP,
                               float* __restrict__ C,
                               int Ntiles) {
    const int mt   = blockIdx.x;
    const int lane = threadIdx.x;     // 0..31
    const int lo   = lane & 15;
    const int hi   = lane >> 4;       // 0 or 1
    const int Np   = Ntiles * 16;

    const _Float16* __restrict__ arow = Ap + (size_t)(mt * 16 + lo) * KPAD;
    const v8h al0 = *(const v8h*)(arow + 8 * hi);        // kstep0 h=0..7
    const v8h ah0 = *(const v8h*)(arow + 16 + 8 * hi);   // kstep0 h=8..15
    const v8h al1 = *(const v8h*)(arow + 32 + 8 * hi);   // kstep1 h=0..7
    const v8h ah1 = *(const v8h*)(arow + 48 + 8 * hi);   // kstep1 h=8..15
    const v16h a0 = __builtin_shufflevector(al0, ah0, 0,1,2,3,4,5,6,7,8,9,10,11,12,13,14,15);
    const v16h a1 = __builtin_shufflevector(al1, ah1, 0,1,2,3,4,5,6,7,8,9,10,11,12,13,14,15);

    for (int nt = 0; nt < Ntiles; ++nt) {
        const _Float16* __restrict__ brow = Bp + (size_t)(nt * 16 + lo) * KPAD;
        const v16h b0 = __builtin_shufflevector(*(const v8h*)(brow + 8 * hi), *(const v8h*)(brow + 16 + 8 * hi), 0,1,2,3,4,5,6,7,8,9,10,11,12,13,14,15);        // kstep0
        const v16h b1 = __builtin_shufflevector(*(const v8h*)(brow + 32 + 8 * hi), *(const v8h*)(brow + 48 + 8 * hi), 0,1,2,3,4,5,6,7,8,9,10,11,12,13,14,15);   // kstep1

        v8f acc = {};
        acc = __builtin_amdgcn_wmma_f32_16x16x32_f16(false, a0, false, b0,
                                                     (short)0, acc, false, false);
        acc = __builtin_amdgcn_wmma_f32_16x16x32_f16(false, a1, false, b1,
                                                     (short)0, acc, false, false);

        const int n  = nt * 16 + lo;
        const float bv = biasP[n];
        float* __restrict__ crow = C + (size_t)(mt * 16 + (hi << 3)) * Np + n;
#pragma unroll
        for (int v = 0; v < 8; ++v) {
            crow[(size_t)v * Np] = acc[v] + bv;   // unconditional, padded C
        }
    }
}

// ---------------------------------------------------------------------------
// Edge phase: msg[e,f] = (P1[dst,f] + P2[src,f] + lin_b[f]) * sigmoid(-attr[e])
// scatter-added into agg[dst,f].  One thread per (edge, feature); feature is
// the fast dimension -> coalesced P1/P2 reads + L2-local fp32 atomics.
// P1/P2 have padded row stride NP_P.
// ---------------------------------------------------------------------------
__global__ void edge_scatter(const int* __restrict__ edge_index, int E,
                             const float* __restrict__ edge_attr,
                             const float* __restrict__ P1,
                             const float* __restrict__ P2,
                             const float* __restrict__ lin_b,
                             float* __restrict__ agg) {
    const int tid = blockIdx.x * blockDim.x + threadIdx.x;
    const int total = E * FEAT;
    if (tid >= total) return;
    const int e = tid / FEAT;
    const int f = tid - e * FEAT;
    const int s = edge_index[e];        // edge_index[0][e] = src
    const int d = edge_index[E + e];    // edge_index[1][e] = dst
    const float gate = 1.0f / (1.0f + __expf(edge_attr[e]));   // sigmoid(-attr)
    const float v = (P1[d * NP_P + f] + P2[s * NP_P + f] + lin_b[f]) * gate;
    __hip_atomic_fetch_add(&agg[d * FEAT + f], v,
                           __ATOMIC_RELAXED, __HIP_MEMORY_SCOPE_AGENT);
}

// ---------------------------------------------------------------------------
// GRU elementwise combine (gi/gh strided NP_G, precomputed by WMMA GEMMs)
// ---------------------------------------------------------------------------
__global__ void gru_combine(const float* __restrict__ gi,
                            const float* __restrict__ gh,
                            const float* __restrict__ agg,
                            float* __restrict__ hx, int n_nodes) {
    const int tid = blockIdx.x * blockDim.x + threadIdx.x;
    if (tid >= n_nodes * FEAT) return;
    const int node = tid / FEAT;
    const int f    = tid - node * FEAT;
    const int b    = node * NP_G;
    const float r  = sigf(gi[b + f]             + gh[b + f]);
    const float z  = sigf(gi[b + FEAT + f]      + gh[b + FEAT + f]);
    const float ng = tanhf(gi[b + 2 * FEAT + f] + r * gh[b + 2 * FEAT + f]);
    hx[tid] = (1.0f - z) * ng + z * agg[tid];
}

// ---------------------------------------------------------------------------
// Serial bidirectional LSTM: 2 blocks (dir 0 = fwd, dir 1 = bwd), 1 wave each.
// Lane j (0..19) holds h[j], c[j] and rows {j, 20+j, 40+j, 60+j} of w_hh in
// registers.  Per step: 20 lane-broadcasts x 4 FMAs (fully unrolled) +
// 5 transcendentals.  xp = hx @ w_ihᵀ + b was precomputed by WMMA GEMM
// (row stride H4 == padded stride, 5 tiles exactly).
// ---------------------------------------------------------------------------
__global__ void lstm_scan(const float* __restrict__ xpf,
                          const float* __restrict__ xpb,
                          const float* __restrict__ whh_f,
                          const float* __restrict__ whh_b,
                          float* __restrict__ hcat, int n) {
    const int dir = blockIdx.x;            // 0 fwd, 1 bwd
    const int j   = threadIdx.x;           // 0..31 (wave32)
    const bool act = (j < HID);
    const int jj = act ? j : 0;
    const float* __restrict__ xp  = dir ? xpb   : xpf;
    const float* __restrict__ whh = dir ? whh_b : whh_f;

    float w0[HID], w1[HID], w2[HID], w3[HID];
#pragma unroll
    for (int k = 0; k < HID; ++k) {
        w0[k] = act ? whh[jj * HID + k]              : 0.0f;
        w1[k] = act ? whh[(HID + jj) * HID + k]      : 0.0f;
        w2[k] = act ? whh[(2 * HID + jj) * HID + k]  : 0.0f;
        w3[k] = act ? whh[(3 * HID + jj) * HID + k]  : 0.0f;
    }

    float h = 0.0f, c = 0.0f;
    for (int t = 0; t < n; ++t) {
        const int idx = dir ? (n - 1 - t) : t;
        const float* __restrict__ xr = xp + idx * H4;
        const int nidx = dir ? (n - 2 - t) : (t + 1);
        if (nidx >= 0 && nidx < n)
            __builtin_prefetch(xp + nidx * H4, 0, 3);   // global_prefetch_b8

        float g0 = act ? xr[jj]            : 0.0f;
        float g1 = act ? xr[HID + jj]      : 0.0f;
        float g2 = act ? xr[2 * HID + jj]  : 0.0f;
        float g3 = act ? xr[3 * HID + jj]  : 0.0f;
#pragma unroll
        for (int k = 0; k < HID; ++k) {
            const float hb = __shfl(h, k, 32);
            g0 = fmaf(w0[k], hb, g0);
            g1 = fmaf(w1[k], hb, g1);
            g2 = fmaf(w2[k], hb, g2);
            g3 = fmaf(w3[k], hb, g3);
        }
        const float ig = sigf(g0);
        const float fg = sigf(g1);
        const float gg = tanhf(g2);
        const float og = sigf(g3);
        c = fg * c + ig * gg;
        h = og * tanhf(c);
        if (act) hcat[idx * (2 * HID) + dir * HID + j] = h;
    }
}

// ---------------------------------------------------------------------------
// Classifier: out[i] = hcat[i,:] . cls_w + cls_b
// ---------------------------------------------------------------------------
__global__ void classifier(const float* __restrict__ hcat,
                           const float* __restrict__ cls_w,
                           const float* __restrict__ cls_b,
                           float* __restrict__ out, int n) {
    const int i = blockIdx.x * blockDim.x + threadIdx.x;
    if (i >= n) return;
    float acc = cls_b[0];
#pragma unroll
    for (int jc = 0; jc < 2 * HID; ++jc)
        acc = fmaf(hcat[i * (2 * HID) + jc], cls_w[jc], acc);
    out[i] = acc;
}

// ---------------------------------------------------------------------------
// Launch
// ---------------------------------------------------------------------------
extern "C" void kernel_launch(void* const* d_in, const int* in_sizes, int n_in,
                              void* d_out, int out_size, void* d_ws, size_t ws_size,
                              hipStream_t stream) {
    const float* x          = (const float*)d_in[0];
    const int*   edge_index = (const int*)  d_in[1];
    const float* edge_attr  = (const float*)d_in[2];
    const float* lin_w      = (const float*)d_in[3];   // (F, 2F) row-major
    const float* lin_b      = (const float*)d_in[4];
    const float* gru_w_ih   = (const float*)d_in[5];   // (3F, F)
    const float* gru_w_hh   = (const float*)d_in[6];
    const float* gru_b_ih   = (const float*)d_in[7];
    const float* gru_b_hh   = (const float*)d_in[8];
    const float* lw_ih_f    = (const float*)d_in[9];   // (4H, F)
    const float* lw_hh_f    = (const float*)d_in[10];  // (4H, H)
    const float* lb_f       = (const float*)d_in[11];
    const float* lw_ih_b    = (const float*)d_in[12];
    const float* lw_hh_b    = (const float*)d_in[13];
    const float* lb_b       = (const float*)d_in[14];
    const float* cls_w      = (const float*)d_in[15];
    const float* cls_b      = (const float*)d_in[16];
    float* out = (float*)d_out;

    const int N = in_sizes[0] / FEAT;   // 20000
    const int E = in_sizes[2];          // 1280000

    const int Mtiles = (N + 15) / 16;
    const int Mp     = Mtiles * 16;

    // ---- Workspace layout (floats first, then f16 packed region) ----------
    float* ws   = (float*)d_ws;
    float* P1   = ws;                        // Mp*NP_P (padded)
    float* P2   = P1 + (size_t)Mp * NP_P;    // Mp*NP_P
    float* agg  = P2 + (size_t)Mp * NP_P;    // N*F (tight, atomic target)
    float* gi   = agg + (size_t)N * FEAT;    // Mp*NP_G (padded)
    float* gh   = gi + (size_t)Mp * NP_G;    // Mp*NP_G
    float* bz   = gh + (size_t)Mp * NP_G;    // 128 zeros   (no-bias pad)
    float* bp0  = bz + 128;                  // 128 padded bias
    float* bp1  = bp0 + 128;                 // 128 padded bias
    float* fend = bp1 + 128;
    float* hx   = P1;                        // reuse (P1 dead after edges)  N*F tight
    float* xpf  = gi;                        // reuse (gi dead after GRU)    Mp*H4
    float* xpb  = gh;                        // reuse                        Mp*H4
    float* hcat = P2;                        // reuse (P2 dead after edges)  N*2H tight

    _Float16* Ap  = (_Float16*)fend;             // Mp*64 halves (reused 3x)
    _Float16* Bp0 = Ap + (size_t)Mp * KPAD;      // 128*64 halves
    _Float16* Bp1 = Bp0 + 128 * KPAD;            // 128*64 halves

    const int packA_n = Mp * KPAD;
    const int packA_g = (packA_n + 255) / 256;

    // agg must start at zero for the atomic scatter; bz is the zero bias pad
    zero_f32<<<(N * FEAT + 255) / 256, 256, 0, stream>>>(agg, N * FEAT);
    zero_f32<<<1, 128, 0, stream>>>(bz, 128);

    // ====== Phase 1: node projections P1 = x@W1ᵀ, P2 = x@W2ᵀ ==============
    pack_f16<<<packA_g, 256, 0, stream>>>(x, FEAT, N, FEAT, Ap, Mp);
    pack_f16<<<(48 * KPAD + 255) / 256, 256, 0, stream>>>(lin_w,        2 * FEAT, FEAT, FEAT, Bp0, 48);
    pack_f16<<<(48 * KPAD + 255) / 256, 256, 0, stream>>>(lin_w + FEAT, 2 * FEAT, FEAT, FEAT, Bp1, 48);
    gemm_wmma_f16p<<<Mtiles, 32, 0, stream>>>(Ap, Bp0, bz, P1, 3);
    gemm_wmma_f16p<<<Mtiles, 32, 0, stream>>>(Ap, Bp1, bz, P2, 3);

    // ====== Phase 2: edge gather + gate + scatter-add ======================
    {
        const int total = E * FEAT;
        edge_scatter<<<(total + 255) / 256, 256, 0, stream>>>(
            edge_index, E, edge_attr, P1, P2, lin_b, agg);
    }

    // ====== Phase 3: GRU ===================================================
    pack_f16<<<packA_g, 256, 0, stream>>>(agg, FEAT, N, FEAT, Ap, Mp);
    pack_f16<<<(128 * KPAD + 255) / 256, 256, 0, stream>>>(gru_w_ih, FEAT, G3, FEAT, Bp0, 128);
    pack_f16<<<(128 * KPAD + 255) / 256, 256, 0, stream>>>(gru_w_hh, FEAT, G3, FEAT, Bp1, 128);
    pack_bias<<<1, 128, 0, stream>>>(gru_b_ih, G3, bp0, NP_G);
    pack_bias<<<1, 128, 0, stream>>>(gru_b_hh, G3, bp1, NP_G);
    gemm_wmma_f16p<<<Mtiles, 32, 0, stream>>>(Ap, Bp0, bp0, gi, 8);
    gemm_wmma_f16p<<<Mtiles, 32, 0, stream>>>(Ap, Bp1, bp1, gh, 8);
    gru_combine<<<(N * FEAT + 255) / 256, 256, 0, stream>>>(gi, gh, agg, hx, N);

    // ====== Phase 4: LSTM input projections ================================
    pack_f16<<<packA_g, 256, 0, stream>>>(hx, FEAT, N, FEAT, Ap, Mp);
    pack_f16<<<(80 * KPAD + 255) / 256, 256, 0, stream>>>(lw_ih_f, FEAT, H4, FEAT, Bp0, 80);
    pack_f16<<<(80 * KPAD + 255) / 256, 256, 0, stream>>>(lw_ih_b, FEAT, H4, FEAT, Bp1, 80);
    pack_bias<<<1, 128, 0, stream>>>(lb_f, H4, bp0, H4);
    pack_bias<<<1, 128, 0, stream>>>(lb_b, H4, bp1, H4);
    gemm_wmma_f16p<<<Mtiles, 32, 0, stream>>>(Ap, Bp0, bp0, xpf, 5);
    gemm_wmma_f16p<<<Mtiles, 32, 0, stream>>>(Ap, Bp1, bp1, xpb, 5);

    // ====== Phase 5: serial recurrences (fwd + bwd concurrently) ===========
    lstm_scan<<<2, 32, 0, stream>>>(xpf, xpb, lw_hh_f, lw_hh_b, hcat, N);

    // ====== Phase 6: classifier ============================================
    classifier<<<(N + 255) / 256, 256, 0, stream>>>(hcat, cls_w, cls_b, out, N);
}
